// GptOssExperts_56083682951827
// MI455X (gfx1250) — compile-verified, matrix operand order checked by reading, not run on hardware
//
#include <hip/hip_runtime.h>
#include <cstdint>

#define E_ 8
#define H_ 1024
#define I_ 2048
#define T_ 4096

typedef __attribute__((ext_vector_type(16))) __bf16 v16bf;
typedef __attribute__((ext_vector_type(8)))  float  v8f;

union Frag16 { v16bf v; uint4 q[2]; };

// ---- CDNA5 async copy global -> LDS (ASYNCcnt-tracked) ----
__device__ __forceinline__ void async_copy_b128(void* lds_ptr, const void* gptr) {
  unsigned lds_off = (unsigned)(uintptr_t)lds_ptr;   // low 32 bits of flat shared addr == LDS offset
  asm volatile("global_load_async_to_lds_b128 %0, %1, off"
               :: "v"(lds_off), "v"(gptr) : "memory");
}
template <int N>
__device__ __forceinline__ void wait_async() {
  asm volatile("s_wait_asynccnt %0" :: "i"(N) : "memory");
}

// Load a 16x32 bf16 A/B fragment from a row-major LDS tile (row stride ldk elems).
// ISA layout: lanes 0-15 -> row, K{0..7,16..23}; lanes 16-31 -> row, K{8..15,24..31}.
__device__ __forceinline__ v16bf load_frag(const __bf16* base, int row0, int hlf, int l16,
                                           int ldk, int kofs) {
  const __bf16* p = base + (size_t)(row0 + l16) * ldk + kofs + hlf * 8;
  Frag16 f;
  f.q[0] = *(const uint4*)(p);
  f.q[1] = *(const uint4*)(p + 16);
  return f.v;
}

// ---------------- prep kernels ----------------
__global__ __launch_bounds__(256) void k_cvt_x(const float* __restrict__ in, __bf16* __restrict__ out) {
  int i = (blockIdx.x * 256 + threadIdx.x) * 4;
  float4 v = *(const float4*)(in + i);
  union { __bf16 b[4]; uint2 u; } pk;
  pk.b[0] = (__bf16)v.x; pk.b[1] = (__bf16)v.y; pk.b[2] = (__bf16)v.z; pk.b[3] = (__bf16)v.w;
  *(uint2*)(out + i) = pk.u;
}

// gate_up_proj (E,H,2I) f32 -> wg[e][i][h], wu[e][i][h] bf16 (de-interleave + transpose)
__global__ __launch_bounds__(256) void k_tr_gateup(const float* __restrict__ gup,
                                                   __bf16* __restrict__ wg,
                                                   __bf16* __restrict__ wu) {
  __shared__ float tg[32][33];
  __shared__ float tu[32][33];
  int e = blockIdx.z;
  int i0 = blockIdx.x * 32;
  int h0 = blockIdx.y * 32;
  int tx = threadIdx.x, ty = threadIdx.y; // 32 x 8
  const float* src = gup + ((size_t)e * H_ + h0) * (2 * I_);
#pragma unroll
  for (int j = 0; j < 4; ++j) {
    int hl = ty + j * 8;
    int ic = i0 + tx;
    tg[hl][tx] = src[(size_t)hl * (2 * I_) + 2 * ic];
    tu[hl][tx] = src[(size_t)hl * (2 * I_) + 2 * ic + 1];
  }
  __syncthreads();
#pragma unroll
  for (int j = 0; j < 4; ++j) {
    int il = ty + j * 8;
    size_t o = ((size_t)e * I_ + (i0 + il)) * H_ + h0 + tx;
    wg[o] = (__bf16)tg[tx][il];
    wu[o] = (__bf16)tu[tx][il];
  }
}

// down_proj (E,I,H) f32 -> wd[e][h][i] bf16 (transpose)
__global__ __launch_bounds__(256) void k_tr_down(const float* __restrict__ dwn, __bf16* __restrict__ wd) {
  __shared__ float t[32][33];
  int e = blockIdx.z;
  int h0 = blockIdx.x * 32;
  int i0 = blockIdx.y * 32;
  int tx = threadIdx.x, ty = threadIdx.y;
  const float* src = dwn + (size_t)e * I_ * H_;
#pragma unroll
  for (int j = 0; j < 4; ++j)
    t[ty + j * 8][tx] = src[(size_t)(i0 + ty + j * 8) * H_ + h0 + tx];
  __syncthreads();
#pragma unroll
  for (int j = 0; j < 4; ++j)
    wd[((size_t)e * H_ + h0 + ty + j * 8) * I_ + i0 + tx] = (__bf16)t[tx][ty + j * 8];
}

// ---------------- GEMM1 + gated activation ----------------
// 128x64 tile, BK=64, triple-buffered async LDS, one barrier per stage.
__global__ __launch_bounds__(256) void k_gemm1_act(const __bf16* __restrict__ x,
                                                   const __bf16* __restrict__ wg,
                                                   const __bf16* __restrict__ wu,
                                                   const float* __restrict__ gub,
                                                   __bf16* __restrict__ act) {
  constexpr int BM = 128, BN = 64, BK = 64, LDK = BK + 8; // 72 elems = 144B rows (16B aligned)
  __shared__ __bf16 As[3][BM * LDK];
  __shared__ __bf16 Bg[3][BN * LDK];
  __shared__ __bf16 Bu[3][BN * LDK];

  const int e  = blockIdx.z;
  const int i0 = blockIdx.x * BN;
  const int t0 = blockIdx.y * BM;
  const int tid = threadIdx.x;
  const int lane = tid & 31, wid = tid >> 5;
  const int hlf = lane >> 4, l16 = lane & 15;
  const int m_wave = (wid & 1) * 64;
  const int n_wave = (wid >> 1) * 16;

  const __bf16* xg  = x  + (size_t)t0 * H_;
  const __bf16* wgg = wg + ((size_t)e * I_ + i0) * H_;
  const __bf16* wug = wu + ((size_t)e * I_ + i0) * H_;

  auto issue = [&](int buf, int k0) {
#pragma unroll
    for (int tc = 0; tc < 4; ++tc) {               // A: 128 rows x 64 cols = 1024 x 16B chunks
      int c = tid + tc * 256;
      int row = c >> 3, seg = c & 7;
      async_copy_b128(&As[buf][row * LDK + seg * 8], xg + (size_t)row * H_ + k0 + seg * 8);
    }
#pragma unroll
    for (int tc = 0; tc < 2; ++tc) {               // Bg/Bu: 64 rows x 64 cols = 512 chunks each
      int c = tid + tc * 256;
      int row = c >> 3, seg = c & 7;
      async_copy_b128(&Bg[buf][row * LDK + seg * 8], wgg + (size_t)row * H_ + k0 + seg * 8);
      async_copy_b128(&Bu[buf][row * LDK + seg * 8], wug + (size_t)row * H_ + k0 + seg * 8);
    }
  };

  v8f accg[4] = {};
  v8f accu[4] = {};

  issue(0, 0);
  issue(1, BK);
  const int KT = H_ / BK; // 16
  for (int kt = 0; kt < KT; ++kt) {
    if (kt + 1 < KT) wait_async<8>();   // my stage-kt loads done (next stage may be in flight)
    else             wait_async<0>();
    __syncthreads();                    // everyone's stage-kt data in LDS; compute(kt-1) done by all
    if (kt + 2 < KT) issue((kt + 2) % 3, (kt + 2) * BK);  // buffer last read at compute(kt-1): safe
    const int buf = kt % 3;
#pragma unroll
    for (int ks = 0; ks < 2; ++ks) {
      v16bf bg = load_frag(&Bg[buf][0], n_wave, hlf, l16, LDK, ks * 32);
      v16bf bu = load_frag(&Bu[buf][0], n_wave, hlf, l16, LDK, ks * 32);
#pragma unroll
      for (int fm = 0; fm < 4; ++fm) {
        v16bf a = load_frag(&As[buf][0], m_wave + fm * 16, hlf, l16, LDK, ks * 32);
        accg[fm] = __builtin_amdgcn_wmma_f32_16x16x32_bf16(false, a, false, bg, (short)0, accg[fm], false, false);
        accu[fm] = __builtin_amdgcn_wmma_f32_16x16x32_bf16(false, a, false, bu, (short)0, accu[fm], false, false);
      }
    }
  }

  const int col = i0 + n_wave + l16;
  const float bgs = gub[(size_t)e * (2 * I_) + 2 * col];
  const float bus = gub[(size_t)e * (2 * I_) + 2 * col + 1];
#pragma unroll
  for (int fm = 0; fm < 4; ++fm) {
#pragma unroll
    for (int r = 0; r < 8; ++r) {
      int trow = t0 + m_wave + fm * 16 + hlf * 8 + r;
      float g = accg[fm][r] + bgs;
      float u = accu[fm][r] + bus;
      g = fminf(g, 7.0f);
      u = fminf(fmaxf(u, -7.0f), 7.0f);
      float glu = g / (1.0f + __expf(-1.702f * g));
      act[((size_t)e * T_ + trow) * I_ + col] = (__bf16)((u + 1.0f) * glu);
    }
  }
}

// ---------------- GEMM2 + routing-weighted reduce over experts ----------------
__global__ __launch_bounds__(256) void k_gemm2_out(const __bf16* __restrict__ act,
                                                   const __bf16* __restrict__ wd,
                                                   const float* __restrict__ dnb,
                                                   const float* __restrict__ rw,
                                                   float* __restrict__ out) {
  constexpr int BM = 128, BN = 64, BK = 64, LDK = BK + 8;
  __shared__ __bf16 As[3][BM * LDK];
  __shared__ __bf16 Bd[3][BN * LDK];

  const int h0 = blockIdx.x * BN;
  const int t0 = blockIdx.y * BM;
  const int tid = threadIdx.x;
  const int lane = tid & 31, wid = tid >> 5;
  const int hlf = lane >> 4, l16 = lane & 15;
  const int m_wave = (wid & 1) * 64;
  const int n_wave = (wid >> 1) * 16;
  const int n = h0 + n_wave + l16;

  v8f oacc[4] = {};

  for (int e = 0; e < E_; ++e) {
    const __bf16* ag  = act + ((size_t)e * T_ + t0) * I_;
    const __bf16* wdg = wd  + ((size_t)e * H_ + h0) * I_;

    auto issue = [&](int buf, int k0) {
#pragma unroll
      for (int tc = 0; tc < 4; ++tc) {             // A: 128x64 = 1024 chunks
        int c = tid + tc * 256;
        int row = c >> 3, seg = c & 7;
        async_copy_b128(&As[buf][row * LDK + seg * 8], ag + (size_t)row * I_ + k0 + seg * 8);
      }
#pragma unroll
      for (int tc = 0; tc < 2; ++tc) {             // B: 64x64 = 512 chunks
        int c = tid + tc * 256;
        int row = c >> 3, seg = c & 7;
        async_copy_b128(&Bd[buf][row * LDK + seg * 8], wdg + (size_t)row * I_ + k0 + seg * 8);
      }
    };

    v8f acc[4] = {};
    issue(0, 0);
    issue(1, BK);
    const int KT = I_ / BK; // 32
    for (int kt = 0; kt < KT; ++kt) {
      if (kt + 1 < KT) wait_async<6>();
      else             wait_async<0>();
      __syncthreads();
      if (kt + 2 < KT) issue((kt + 2) % 3, (kt + 2) * BK);
      const int buf = kt % 3;
#pragma unroll
      for (int ks = 0; ks < 2; ++ks) {
        v16bf b = load_frag(&Bd[buf][0], n_wave, hlf, l16, LDK, ks * 32);
#pragma unroll
        for (int fm = 0; fm < 4; ++fm) {
          v16bf a = load_frag(&As[buf][0], m_wave + fm * 16, hlf, l16, LDK, ks * 32);
          acc[fm] = __builtin_amdgcn_wmma_f32_16x16x32_bf16(false, a, false, b, (short)0, acc[fm], false, false);
        }
      }
    }

    const float bias = dnb[(size_t)e * H_ + n];
#pragma unroll
    for (int fm = 0; fm < 4; ++fm) {
#pragma unroll
      for (int r = 0; r < 8; ++r) {
        int trow = t0 + m_wave + fm * 16 + hlf * 8 + r;
        float w = rw[(size_t)trow * E_ + e];
        oacc[fm][r] += w * (acc[fm][r] + bias);
      }
    }
    __syncthreads();   // protect buffer reuse across the expert boundary
  }

#pragma unroll
  for (int fm = 0; fm < 4; ++fm) {
#pragma unroll
    for (int r = 0; r < 8; ++r) {
      int trow = t0 + m_wave + fm * 16 + hlf * 8 + r;
      out[(size_t)trow * H_ + n] = oacc[fm][r];
    }
  }
}

extern "C" void kernel_launch(void* const* d_in, const int* in_sizes, int n_in,
                              void* d_out, int out_size, void* d_ws, size_t ws_size,
                              hipStream_t stream) {
  (void)in_sizes; (void)n_in; (void)out_size; (void)ws_size;
  const float* hs  = (const float*)d_in[0];  // (B,S,H)   = (T,H)
  const float* rw  = (const float*)d_in[1];  // (T,E)
  const float* gup = (const float*)d_in[2];  // (E,H,2I)
  const float* gub = (const float*)d_in[3];  // (E,2I)
  const float* dwn = (const float*)d_in[4];  // (E,I,H)
  const float* dnb = (const float*)d_in[5];  // (E,H)
  float* out = (float*)d_out;                // (T,H)

  __bf16* x_bf = (__bf16*)d_ws;
  __bf16* wg   = x_bf + (size_t)T_ * H_;
  __bf16* wu   = wg   + (size_t)E_ * I_ * H_;
  __bf16* wd   = wu   + (size_t)E_ * I_ * H_;
  __bf16* act  = wd   + (size_t)E_ * H_ * I_;

  k_cvt_x    <<<(T_ * H_) / 1024, 256, 0, stream>>>(hs, x_bf);
  k_tr_gateup<<<dim3(I_ / 32, H_ / 32, E_), dim3(32, 8), 0, stream>>>(gup, wg, wu);
  k_tr_down  <<<dim3(H_ / 32, I_ / 32, E_), dim3(32, 8), 0, stream>>>(dwn, wd);
  k_gemm1_act<<<dim3(I_ / 64, T_ / 128, E_), 256, 0, stream>>>(x_bf, wg, wu, gub, act);
  k_gemm2_out<<<dim3(H_ / 64, T_ / 128), 256, 0, stream>>>(act, wd, dnb, rw, out);
}
